// VectorQuantizer_13262859010396
// MI455X (gfx1250) — matrix operand order repaired and checked by reading, not compiled, not run on hardware
//
#include <hip/hip_runtime.h>

#define NE    2048
#define ED    128
#define NTOK  131072

typedef _Float16 v4h  __attribute__((ext_vector_type(4)));
typedef _Float16 v8h  __attribute__((ext_vector_type(8)));
typedef _Float16 v16h __attribute__((ext_vector_type(16)));
typedef float    v4f  __attribute__((ext_vector_type(4)));
typedef float    v8f  __attribute__((ext_vector_type(8)));
typedef unsigned int u32x4 __attribute__((ext_vector_type(4)));
typedef int          i32x4 __attribute__((ext_vector_type(4)));
typedef int          i32x8 __attribute__((ext_vector_type(8)));

// LDS weight tile: 128 codes x 128 dims f16, row stride 136 halves (272B).
// 272B = 68 banks -> consecutive rows rotate by 4 banks -> the 16 lanes of a
// B-fragment ds_load_b128 hit disjoint banks. The TDM reproduces this stride
// via pad_interval=64 DWORDs (256B row) + pad_amount=4 DWORDs (16B).
#define LROW 136
#define TILE_BYTES (128 * LROW * 2)   // 34816 B per buffer

// ---------------------------------------------------------------------------
// TDM: issue one tensor_load_to_lds moving a 128x128 f16 tile (2D descriptor).
// ---------------------------------------------------------------------------
__device__ __forceinline__ void tdm_load_tile(const _Float16* w16, int nbase,
                                              unsigned lds_addr) {
  const unsigned long long ga =
      (unsigned long long)(size_t)(w16 + (size_t)nbase * ED);
  u32x4 g0;
  g0[0] = 1u;                                   // count=1, user descriptor
  g0[1] = lds_addr;                             // LDS byte address of tile
  g0[2] = (unsigned)(ga & 0xffffffffu);         // global_addr[31:0]
  g0[3] = (unsigned)((ga >> 32) & 0x01ffffffu)  // global_addr[56:32]
          | (2u << 30);                         // type = 2 ("image")
  i32x8 g1;
  g1[0] = (1 << 16)      // data_size = 1 -> 2 bytes
        | (1 << 20)      // pad_enable
        | (5 << 22)      // pad_interval: 64 DWORDs (one 256B tile row)
        | (3 << 25);     // pad_amount: 4 DWORDs (16B) -> 272B LDS stride
  g1[1] = (int)(128u << 16);  // tensor_dim0[15:0] = 128 (bits 63:48)
  g1[2] = (int)(128u << 16);  // tensor_dim0 hi = 0; tensor_dim1[15:0] = 128
  g1[3] = (int)(128u << 16);  // tensor_dim1 hi = 0; tile_dim0 = 128
  g1[4] = 128;                // tile_dim1 = 128; tile_dim2 = 0
  g1[5] = 128;                // tensor_dim0_stride[31:0] = 128 elements
  g1[6] = 0;                  // stride0 hi / stride1 lo
  g1[7] = 0;                  // stride1 hi
  const i32x4 gz = {0, 0, 0, 0};
#if defined(__clang_major__) && __clang_major__ >= 23
  const i32x8 gz8 = {0, 0, 0, 0, 0, 0, 0, 0};
  __builtin_amdgcn_tensor_load_to_lds(g0, g1, gz, gz, gz8, 0);
#else
  __builtin_amdgcn_tensor_load_to_lds(g0, g1, gz, gz, 0);
#endif
}

// ---------------------------------------------------------------------------
// Phase 0: codebook f32 -> f16 copy (TDM source) + wnh[n] = 0.5*||w[n]||^2
// ---------------------------------------------------------------------------
__global__ __launch_bounds__(256) void vq_prep_kernel(
    const float* __restrict__ w, _Float16* __restrict__ w16,
    float* __restrict__ wnh) {
  const int gtid = blockIdx.x * 256 + threadIdx.x;  // grid 64 -> 16384 threads
  for (int i = gtid; i < NE * ED / 4; i += 16384) {
    v4f v = *(const v4f*)(w + (size_t)i * 4);
    *(v4h*)(w16 + (size_t)i * 4) = __builtin_convertvector(v, v4h);
  }
  const int gwave = gtid >> 5;  // 512 waves, 4 codes each
  const int lane  = threadIdx.x & 31;
  for (int code = gwave; code < NE; code += 512) {
    v4f v = *(const v4f*)(w + (size_t)code * ED + lane * 4);
    float s = v[0]*v[0] + v[1]*v[1] + v[2]*v[2] + v[3]*v[3];
    for (int t = 16; t >= 1; t >>= 1) s += __shfl_xor(s, t, 32);
    if (lane == 0) wnh[code] = 0.5f * s;
  }
}

// ---------------------------------------------------------------------------
// Phase 1: argmax over codes of (x.w - 0.5||w||^2) via V_WMMA_F32_16X16X32_F16.
// Block = 256 threads = 8 waves, 128 token rows; A fragments in registers;
// codebook tiles DMA'd by the TDM into double-buffered LDS.
// ---------------------------------------------------------------------------
__global__ __launch_bounds__(256) void vq_argmin_kernel(
    const float* __restrict__ x, const _Float16* __restrict__ w16,
    const float* __restrict__ wnh, float* __restrict__ idx_out) {
  __shared__ _Float16 lw[2 * 128 * LROW];

  const int tid  = threadIdx.x;
  const int wave = tid >> 5;
  const int lane = tid & 31;
  const int h    = lane >> 4;   // half-wave select (K-group / M-group)
  const int ln   = lane & 15;
  const unsigned lds_base = (unsigned)(size_t)(&lw[0]);

  // ---- This wave's 16 x-rows into WMMA 16-bit A layout (f32 -> f16). ----
  // A 16x32 f16 layout: lane l holds M=l%16; K = {8h..8h+7} U {16+8h..16+8h+7}
  const int mrow = blockIdx.x * 128 + wave * 16 + ln;
  const float* xr = x + (size_t)mrow * ED;
  v16h a[4];
#pragma unroll
  for (int s = 0; s < 4; ++s) {
    v8f f0 = *(const v8f*)(xr + 32 * s + 8 * h);
    v8f f1 = *(const v8f*)(xr + 32 * s + 16 + 8 * h);
    v8h h0 = __builtin_convertvector(f0, v8h);
    v8h h1 = __builtin_convertvector(f1, v8h);
    a[s] = __builtin_shufflevector(h0, h1, 0, 1, 2, 3, 4, 5, 6, 7,
                                   8, 9, 10, 11, 12, 13, 14, 15);
  }

  float best[8];
  int   bidx[8];
#pragma unroll
  for (int r = 0; r < 8; ++r) { best[r] = -3.0e38f; bidx[r] = 0; }

  // Prologue: DMA tile 0 into buffer 0.
  if (wave == 0) {
    tdm_load_tile(w16, 0, lds_base);
    __builtin_amdgcn_s_wait_tensorcnt((short)0);
  }
  __syncthreads();

  for (int tile = 0; tile < NE / 128; ++tile) {
    // Prefetch next tile into the other buffer (free: barrier ending the
    // previous iteration guarantees nobody still reads that buffer).
    if (wave == 0 && tile + 1 < NE / 128)
      tdm_load_tile(w16, (tile + 1) * 128,
                    lds_base + ((tile + 1) & 1) * TILE_BYTES);

    const _Float16* buf = &lw[(tile & 1) * (128 * LROW)];
    const int nbase = tile * 128;

    // ---- 8 N-chunks of 16 codes; K=128 as 4 chained WMMAs per chunk. ----
    for (int c = 0; c < 8; ++c) {
      const int nloc = c * 16 + ln;  // this lane's B column (code row)
      v8f acc = {};
#pragma unroll
      for (int s = 0; s < 4; ++s) {
        // B 32x16 f16 layout: lane l holds N=l%16; K = 32s + {8h..8h+7} U {16+8h..16+8h+7}
        const _Float16* bp = &buf[nloc * LROW + s * 32 + h * 8];
        v8h b0 = *(const v8h*)bp;
        v8h b1 = *(const v8h*)(bp + 16);
        v16h b = __builtin_shufflevector(b0, b1, 0, 1, 2, 3, 4, 5, 6, 7,
                                         8, 9, 10, 11, 12, 13, 14, 15);
        acc = __builtin_amdgcn_wmma_f32_16x16x32_f16(
            false, a[s], false, b, (short)0, acc, false, false);
      }
      const float wn = wnh[nbase + nloc];  // L2-resident broadcast-ish load
      const int   n  = nbase + nloc;
#pragma unroll
      for (int r = 0; r < 8; ++r) {
        const float v = acc[r] - wn;  // maximize x.w - 0.5||w||^2
        if (v > best[r]) { best[r] = v; bidx[r] = n; }
      }
    }

    if (wave == 0) __builtin_amdgcn_s_wait_tensorcnt((short)0);
    __syncthreads();
  }

  // ---- Reduce across the 16 lanes holding each row (masks 1,2,4,8 stay
  //      within each half-wave); tie-break to smaller index (argmin). ----
#pragma unroll
  for (int r = 0; r < 8; ++r) {
    float bv = best[r];
    int   bi = bidx[r];
    for (int t = 1; t < 16; t <<= 1) {
      const float ov = __shfl_xor(bv, t, 32);
      const int   oi = __shfl_xor(bi, t, 32);
      if (ov > bv || (ov == bv && oi < bi)) { bv = ov; bi = oi; }
    }
    if (ln == 0) {
      const int row = blockIdx.x * 128 + wave * 16 + h * 8 + r;
      idx_out[row] = (float)bi;
    }
  }
}

// ---------------------------------------------------------------------------
// Phase 2: x_q = weight[idx]; per-block partial sum of (x_q - x)^2.
// ---------------------------------------------------------------------------
__global__ __launch_bounds__(256) void vq_gather_kernel(
    const float* __restrict__ x, const float* __restrict__ w,
    const float* __restrict__ idxf, float* __restrict__ xq,
    float* __restrict__ partials) {
  __shared__ float sred[8];
  const int wave = threadIdx.x >> 5, lane = threadIdx.x & 31;
  const int base = blockIdx.x * 256;
  float accum = 0.0f;
  for (int it = 0; it < 32; ++it) {
    const int tok  = base + it * 8 + wave;
    const int code = (int)idxf[tok];
    v4f wv = *(const v4f*)(w + (size_t)code * ED + lane * 4);
    v4f xv = *(const v4f*)(x + (size_t)tok * ED + lane * 4);
    *(v4f*)(xq + (size_t)tok * ED + lane * 4) = wv;
    const v4f d = wv - xv;
    accum = fmaf(d[0], d[0],
            fmaf(d[1], d[1], fmaf(d[2], d[2], fmaf(d[3], d[3], accum))));
  }
  for (int t = 16; t >= 1; t >>= 1) accum += __shfl_xor(accum, t, 32);
  if (lane == 0) sred[wave] = accum;
  __syncthreads();
  if (threadIdx.x == 0) {
    float s = 0.0f;
    for (int i = 0; i < 8; ++i) s += sred[i];
    partials[blockIdx.x] = s;  // plain store: deterministic
  }
}

// ---------------------------------------------------------------------------
// Phase 3: loss = 1.25 * mean((x_q - x)^2) from 512 partials.
// ---------------------------------------------------------------------------
__global__ __launch_bounds__(256) void vq_loss_kernel(
    const float* __restrict__ partials, float* __restrict__ loss_out) {
  __shared__ float sred[8];
  float s = 0.0f;
  for (int i = threadIdx.x; i < NTOK / 256; i += 256) s += partials[i];
  for (int t = 16; t >= 1; t >>= 1) s += __shfl_xor(s, t, 32);
  const int wave = threadIdx.x >> 5, lane = threadIdx.x & 31;
  if (lane == 0) sred[wave] = s;
  __syncthreads();
  if (threadIdx.x == 0) {
    float tot = 0.0f;
    for (int i = 0; i < 8; ++i) tot += sred[i];
    *loss_out = tot * (1.25f / (float)((size_t)NTOK * ED));
  }
}

// ---------------------------------------------------------------------------
extern "C" void kernel_launch(void* const* d_in, const int* in_sizes, int n_in,
                              void* d_out, int out_size, void* d_ws,
                              size_t ws_size, hipStream_t stream) {
  const float* x = (const float*)d_in[0];
  const float* w = (const float*)d_in[1];

  float* out  = (float*)d_out;
  float* xq   = out;                        // NTOK*ED floats
  float* loss = out + (size_t)NTOK * ED;    // 1 float
  float* idxf = loss + 1;                   // NTOK floats (indices as f32)

  _Float16* w16     = (_Float16*)d_ws;               // NE*ED halves (512KB)
  float*    wnh     = (float*)(w16 + (size_t)NE*ED); // 2048 floats
  float*    partials = wnh + NE;                     // 512 floats

  vq_prep_kernel<<<64, 256, 0, stream>>>(w, w16, wnh);
  vq_argmin_kernel<<<NTOK / 128, 256, 0, stream>>>(x, w16, wnh, idxf);
  vq_gather_kernel<<<NTOK / 256, 256, 0, stream>>>(x, w, idxf, xq, partials);
  vq_loss_kernel<<<1, 256, 0, stream>>>(partials, loss);
}
